// GATNetave_5566277616455
// MI455X (gfx1250) — matrix operand order repaired and checked by reading, not compiled, not run on hardware
//
#include <hip/hip_runtime.h>
#include <hip/hip_bf16.h>
#include <stdint.h>

// ---------------------------------------------------------------------------
// GAT 2-layer network for MI455X (gfx1250, wave32).
// GEMMs via v_wmma_f32_16x16x32_bf16 with b128 fragment loads (transposed W);
// edge softmax/aggregation in f32 (L2-resident, atomic-bound).
// ---------------------------------------------------------------------------

typedef __attribute__((ext_vector_type(16))) __bf16 v16bf;
typedef __attribute__((ext_vector_type(8)))  __bf16 v8bf;
typedef __attribute__((ext_vector_type(8)))  float  v8f;

#define N_NODES   10000
#define F_IN      128
#define HID       64
#define HEADS     4
#define NOUT      256          // HEADS*HID
#define E_EDGES   160000
#define E_TOT     170000       // + N self loops
#define NEG_SLOPE 0.2f

static __device__ __forceinline__ __bf16 to_bf16(float x) {
  unsigned u = __float_as_uint(x);
  unsigned r = (u + 0x7fffu + ((u >> 16) & 1u)) >> 16;   // round-nearest-even
  unsigned short h = (unsigned short)r;
  return __builtin_bit_cast(__bf16, h);
}

// order-preserving float -> uint key for atomicMax on signed floats
static __device__ __forceinline__ unsigned fkey(float x) {
  unsigned u = __float_as_uint(x);
  return (u & 0x80000000u) ? ~u : (u | 0x80000000u);
}
static __device__ __forceinline__ float funkey(unsigned k) {
  return (k & 0x80000000u) ? __uint_as_float(k & 0x7fffffffu)
                           : __uint_as_float(~k);
}

// ---------------------------------------------------------------------------
__global__ void k_zero_u32(unsigned* p, int n) {
  int i = blockIdx.x * blockDim.x + threadIdx.x;
  int stride = gridDim.x * blockDim.x;
  for (; i < n; i += stride) p[i] = 0u;
}

__global__ void k_f32_to_bf16(const float* __restrict__ in, __bf16* __restrict__ out, int n) {
  int i = blockIdx.x * blockDim.x + threadIdx.x;
  int stride = gridDim.x * blockDim.x;
  for (; i < n; i += stride) out[i] = to_bf16(in[i]);
}

// W[K,256] f32 -> Wt[256,K] bf16 (transpose so B fragments are K-contiguous)
__global__ void k_w_to_bf16t(const float* __restrict__ W, __bf16* __restrict__ Wt, int K) {
  int i = blockIdx.x * blockDim.x + threadIdx.x;     // over K*256, coalesced read
  if (i >= K * NOUT) return;
  int k = i >> 8, n = i & 255;
  Wt[n * K + k] = to_bf16(W[i]);
}

// C[M x 256] = A[M x K] * Bt^T, Bt stored [256 x K]; one wave per 16x16 C tile.
// Fragment layouts per CDNA5 ISA 7.12.2; all fragment loads are 16B vectors.
template <int K>
__global__ void k_gemm_wmma(const __bf16* __restrict__ A,
                            const __bf16* __restrict__ Bt,
                            float* __restrict__ C, int M) {
  int wave = (int)((blockIdx.x * blockDim.x + threadIdx.x) >> 5);
  int lane = threadIdx.x & 31;
  int tile_m = wave >> 4;          // 256/16 = 16 N-tiles
  int tile_n = wave & 15;
  if (tile_m * 16 >= M) return;    // wave-uniform guard: EXEC all-ones at WMMA

  int m = tile_m * 16 + (lane & 15);
  int n = tile_n * 16 + (lane & 15);
  int kbA = (lane & 16) ? 8 : 0;   // A: lane halves split K 0-7/8-15 (+16 for hi quad)
  int kbB = (lane & 16) ? 8 : 0;   // B: lane halves split K 0-7/8-15 (+16 for hi quad)

  const __bf16* arow = A  + (size_t)m * K;
  const __bf16* brow = Bt + (size_t)n * K;

  v8f c = {};
#pragma unroll
  for (int k0 = 0; k0 < K; k0 += 32) {
    v8bf alo = *(const v8bf*)(arow + k0 + kbA);        // K = k0+kbA   .. +7
    v8bf ahi = *(const v8bf*)(arow + k0 + kbA + 16);   // K = k0+kbA+16.. +23
    v16bf a = __builtin_shufflevector(alo, ahi,
        0, 1, 2, 3, 4, 5, 6, 7, 8, 9, 10, 11, 12, 13, 14, 15);
    v8bf blo = *(const v8bf*)(brow + k0 + kbB);        // K = k0+kbB   .. +7
    v8bf bhi = *(const v8bf*)(brow + k0 + kbB + 16);   // K = k0+kbB+16.. +23
    v16bf b = __builtin_shufflevector(blo, bhi,
        0, 1, 2, 3, 4, 5, 6, 7, 8, 9, 10, 11, 12, 13, 14, 15);
    c = __builtin_amdgcn_wmma_f32_16x16x32_bf16(
        /*neg_a=*/false, a, /*neg_b=*/false, b,
        /*c_mod=*/(short)0, c, /*reuse_a=*/false, /*reuse_b=*/false);
  }
  int mbase = tile_m * 16 + ((lane & 16) ? 8 : 0);     // C/D: VGPR r -> M = mbase + r
#pragma unroll
  for (int r = 0; r < 8; ++r)
    C[(size_t)(mbase + r) * NOUT + n] = c[r];
}

// al_src[n,h] = <h[n,h,:], a_src[h,:]>, al_dst likewise
__global__ void k_attn_logits(const float* __restrict__ hf,
                              const float* __restrict__ a_src,
                              const float* __restrict__ a_dst,
                              float* __restrict__ als, float* __restrict__ ald) {
  int i = blockIdx.x * blockDim.x + threadIdx.x;       // i in [0, N*HEADS)
  if (i >= N_NODES * HEADS) return;
  int n = i >> 2, h = i & 3;
  const float* hp = hf + (size_t)n * NOUT + h * HID;
  const float* as = a_src + h * HID;
  const float* ad = a_dst + h * HID;
  float s1 = 0.f, s2 = 0.f;
#pragma unroll 8
  for (int c = 0; c < HID; ++c) { float v = hp[c]; s1 += v * as[c]; s2 += v * ad[c]; }
  als[i] = s1;
  ald[i] = s2;
}

static __device__ __forceinline__ void edge_sd(const int* es, const int* ed, int e, int& s, int& d) {
  if (e < E_EDGES) { s = es[e]; d = ed[e]; }
  else             { s = d = e - E_EDGES; }            // appended self loops
}

static __device__ __forceinline__ float leaky(float v) {
  return (v > 0.f) ? v : NEG_SLOPE * v;
}

__global__ void k_edge_max(const int* __restrict__ es, const int* __restrict__ ed,
                           const float* __restrict__ als, const float* __restrict__ ald,
                           unsigned* __restrict__ mkey) {
  int e = blockIdx.x * blockDim.x + threadIdx.x;
  if (e >= E_TOT) return;
  int s, d; edge_sd(es, ed, e, s, d);
  float4 a = *(const float4*)(als + (size_t)s * HEADS);
  float4 b = *(const float4*)(ald + (size_t)d * HEADS);
  atomicMax(&mkey[d * HEADS + 0], fkey(leaky(a.x + b.x)));
  atomicMax(&mkey[d * HEADS + 1], fkey(leaky(a.y + b.y)));
  atomicMax(&mkey[d * HEADS + 2], fkey(leaky(a.z + b.z)));
  atomicMax(&mkey[d * HEADS + 3], fkey(leaky(a.w + b.w)));
}

__global__ void k_edge_exp(const int* __restrict__ es, const int* __restrict__ ed,
                           const float* __restrict__ als, const float* __restrict__ ald,
                           const unsigned* __restrict__ mkey,
                           float* __restrict__ e2, float* __restrict__ ssum) {
  int e = blockIdx.x * blockDim.x + threadIdx.x;
  if (e >= E_TOT) return;
  int s, d; edge_sd(es, ed, e, s, d);
  float4 a = *(const float4*)(als + (size_t)s * HEADS);
  float4 b = *(const float4*)(ald + (size_t)d * HEADS);
  const uint4 mk = *(const uint4*)(mkey + (size_t)d * HEADS);
  float4 ex;
  ex.x = __expf(leaky(a.x + b.x) - funkey(mk.x));
  ex.y = __expf(leaky(a.y + b.y) - funkey(mk.y));
  ex.z = __expf(leaky(a.z + b.z) - funkey(mk.z));
  ex.w = __expf(leaky(a.w + b.w) - funkey(mk.w));
  *(float4*)(e2 + (size_t)e * HEADS) = ex;
  atomicAdd(&ssum[d * HEADS + 0], ex.x);
  atomicAdd(&ssum[d * HEADS + 1], ex.y);
  atomicAdd(&ssum[d * HEADS + 2], ex.z);
  atomicAdd(&ssum[d * HEADS + 3], ex.w);
}

// acc[dst, h, c] += e2[e,h] * hfeat[src, h, c]; 256 threads = one (h,c) lane each
__global__ void k_edge_agg(const int* __restrict__ es, const int* __restrict__ ed,
                           const float* __restrict__ e2,
                           const float* __restrict__ hf, float* __restrict__ acc) {
  int e = blockIdx.x;
  int t = threadIdx.x;                 // 0..255
  int s, d; edge_sd(es, ed, e, s, d);
  float w = e2[(size_t)e * HEADS + (t >> 6)];
  atomicAdd(&acc[(size_t)d * NOUT + t], w * hf[(size_t)s * NOUT + t]);
}

// out[n,c] = mean_h(acc[n,h,c]/(s[n,h]+eps)) + bias[c]; optional ELU + bf16 copy
__global__ void k_finalize(const float* __restrict__ acc, const float* __restrict__ ssum,
                           const float* __restrict__ bias,
                           float* __restrict__ outp, __bf16* __restrict__ outbf,
                           int apply_elu) {
  int i = blockIdx.x * blockDim.x + threadIdx.x;
  if (i >= N_NODES * HID) return;
  int n = i >> 6, c = i & 63;
  const float4 sv = *(const float4*)(ssum + (size_t)n * HEADS);
  float r = acc[(size_t)n * NOUT + 0 * HID + c] / (sv.x + 1e-16f)
          + acc[(size_t)n * NOUT + 1 * HID + c] / (sv.y + 1e-16f)
          + acc[(size_t)n * NOUT + 2 * HID + c] / (sv.z + 1e-16f)
          + acc[(size_t)n * NOUT + 3 * HID + c] / (sv.w + 1e-16f);
  r = r * (1.0f / HEADS) + bias[c];
  if (apply_elu) r = (r > 0.f) ? r : (__expf(r) - 1.f);
  outp[i] = r;
  if (outbf) outbf[i] = to_bf16(r);
}

// ---------------------------------------------------------------------------
// Workspace layout (bytes, all 256B-aligned), reused across the 4 batches
constexpr size_t OFF_XBF   = 0;                         // N*F_IN  bf16
constexpr size_t OFF_H     = OFF_XBF   + 2560000;       // N*256   f32
constexpr size_t OFF_ACC   = OFF_H     + 10240000;      // N*256   f32
constexpr size_t OFF_ALS   = OFF_ACC   + 10240000;      // N*4     f32
constexpr size_t OFF_ALD   = OFF_ALS   + 160000;
constexpr size_t OFF_MKEY  = OFF_ALD   + 160000;        // N*4     u32
constexpr size_t OFF_SSUM  = OFF_MKEY  + 160000;        // N*4     f32
constexpr size_t OFF_E2    = OFF_SSUM  + 160000;        // E_TOT*4 f32
constexpr size_t OFF_MID   = OFF_E2    + 2720000;       // N*64    f32
constexpr size_t OFF_MIDBF = OFF_MID   + 2560000;       // N*64    bf16
constexpr size_t OFF_W1BF  = OFF_MIDBF + 1280000;       // [256,128] bf16 (transposed)
constexpr size_t OFF_W2BF  = OFF_W1BF  + 65536;         // [256,64]  bf16 (transposed)
constexpr size_t WS_NEED   = OFF_W2BF  + 32768;         // ~30.4 MB

static inline int cdiv(int a, int b) { return (a + b - 1) / b; }

extern "C" void kernel_launch(void* const* d_in, const int* in_sizes, int n_in,
                              void* d_out, int out_size, void* d_ws, size_t ws_size,
                              hipStream_t stream) {
  if (ws_size < WS_NEED) return;

  const float* x     = (const float*)d_in[0];   // [4,10000,128]
  const int*   ei    = (const int*)  d_in[1];   // [2,160000]
  const float* W1    = (const float*)d_in[2];   // [128,256]
  const float* asrc1 = (const float*)d_in[3];   // [4,64]
  const float* adst1 = (const float*)d_in[4];
  const float* b1    = (const float*)d_in[5];   // [64]
  const float* W2    = (const float*)d_in[6];   // [64,256]
  const float* asrc2 = (const float*)d_in[7];
  const float* adst2 = (const float*)d_in[8];
  const float* b2    = (const float*)d_in[9];
  float* dout = (float*)d_out;                  // [4,10000,64]

  char* ws = (char*)d_ws;
  __bf16*   xbf   = (__bf16*)  (ws + OFF_XBF);
  float*    hfeat = (float*)   (ws + OFF_H);
  float*    acc   = (float*)   (ws + OFF_ACC);
  float*    als   = (float*)   (ws + OFF_ALS);
  float*    ald   = (float*)   (ws + OFF_ALD);
  unsigned* mkey  = (unsigned*)(ws + OFF_MKEY);
  float*    ssum  = (float*)   (ws + OFF_SSUM);
  float*    e2    = (float*)   (ws + OFF_E2);
  float*    mid   = (float*)   (ws + OFF_MID);
  __bf16*   midbf = (__bf16*)  (ws + OFF_MIDBF);
  __bf16*   w1t   = (__bf16*)  (ws + OFF_W1BF);
  __bf16*   w2t   = (__bf16*)  (ws + OFF_W2BF);

  const int* es = ei;                // src row
  const int* ed = ei + E_EDGES;      // dst row

  const int BLK = 256;
  // weights -> bf16, transposed to [N=256, K] (once per launch)
  k_w_to_bf16t<<<cdiv(F_IN * NOUT, BLK), BLK, 0, stream>>>(W1, w1t, F_IN);
  k_w_to_bf16t<<<cdiv(HID  * NOUT, BLK), BLK, 0, stream>>>(W2, w2t, HID);

  const int gemm_blocks = (625 * 16) / 8;            // 10000 waves, 8 waves/block
  const int edge_blocks = cdiv(E_TOT, BLK);
  const int node_blocks = cdiv(N_NODES * HID, BLK);
  const int nh_blocks   = cdiv(N_NODES * HEADS, BLK);

  for (int b = 0; b < 4; ++b) {
    const float* xb = x + (size_t)b * N_NODES * F_IN;

    // ---- layer 1 ----
    k_f32_to_bf16<<<2048, BLK, 0, stream>>>(xb, xbf, N_NODES * F_IN);
    k_zero_u32<<<2048, BLK, 0, stream>>>((unsigned*)acc, N_NODES * NOUT);
    k_zero_u32<<<cdiv(2 * N_NODES * HEADS, BLK), BLK, 0, stream>>>(mkey, 2 * N_NODES * HEADS); // mkey+ssum
    k_gemm_wmma<F_IN><<<gemm_blocks, BLK, 0, stream>>>(xbf, w1t, hfeat, N_NODES);
    k_attn_logits<<<nh_blocks, BLK, 0, stream>>>(hfeat, asrc1, adst1, als, ald);
    k_edge_max<<<edge_blocks, BLK, 0, stream>>>(es, ed, als, ald, mkey);
    k_edge_exp<<<edge_blocks, BLK, 0, stream>>>(es, ed, als, ald, mkey, e2, ssum);
    k_edge_agg<<<E_TOT, BLK, 0, stream>>>(es, ed, e2, hfeat, acc);
    k_finalize<<<node_blocks, BLK, 0, stream>>>(acc, ssum, b1, mid, midbf, /*elu=*/1);

    // ---- layer 2 ----
    k_zero_u32<<<2048, BLK, 0, stream>>>((unsigned*)acc, N_NODES * NOUT);
    k_zero_u32<<<cdiv(2 * N_NODES * HEADS, BLK), BLK, 0, stream>>>(mkey, 2 * N_NODES * HEADS);
    k_gemm_wmma<HID><<<gemm_blocks, BLK, 0, stream>>>(midbf, w2t, hfeat, N_NODES);
    k_attn_logits<<<nh_blocks, BLK, 0, stream>>>(hfeat, asrc2, adst2, als, ald);
    k_edge_max<<<edge_blocks, BLK, 0, stream>>>(es, ed, als, ald, mkey);
    k_edge_exp<<<edge_blocks, BLK, 0, stream>>>(es, ed, als, ald, mkey, e2, ssum);
    k_edge_agg<<<E_TOT, BLK, 0, stream>>>(es, ed, e2, hfeat, acc);
    k_finalize<<<node_blocks, BLK, 0, stream>>>(acc, ssum, b2,
                                                dout + (size_t)b * N_NODES * HID,
                                                (__bf16*)nullptr, /*elu=*/0);
  }
}
